// GaussianOccHead_77584289235267
// MI455X (gfx1250) — matrix operand order repaired and checked by reading, not compiled, not run on hardware
//
#include <hip/hip_runtime.h>
#include <hip/hip_bf16.h>
#include <math.h>

#define N_POINTS  16384
#define N_GAUSS   3200
#define N_CLASSES 14
#define H_STRIDE  12   // floats per gaussian record (10 used, padded for float4 alignment)

typedef __attribute__((ext_vector_type(16))) _Float16 v16h;
typedef __attribute__((ext_vector_type(8)))  float    v8f;

// ---------------------------------------------------------------------------
// Pass 1: per-gaussian precompute.
//   h[0..9]  = -0.5*log2(e) * [A00, A11, A22, 2A01, 2A02, 2A12, -2(Aμ)x, -2(Aμ)y, -2(Aμ)z, μᵀAμ]
//   h[9]    += log2(opacity)          =>  w(point) = exp2( h · f(point) )
//   ST[c][g] = f16(semantics[g][c]) for c<14, 0 for c=14,15 (B-matrix, transposed, padded)
// ---------------------------------------------------------------------------
__global__ __launch_bounds__(256) void gauss_prep_kernel(
    const float* __restrict__ means, const float* __restrict__ scales,
    const float* __restrict__ rot,   const float* __restrict__ opac,
    const float* __restrict__ sem,
    float* __restrict__ Hg, _Float16* __restrict__ ST)
{
  int g = blockIdx.x * blockDim.x + threadIdx.x;
  if (g >= N_GAUSS) return;

  float qw = rot[g*4+0], qx = rot[g*4+1], qy = rot[g*4+2], qz = rot[g*4+3];
  float inv = rsqrtf(qw*qw + qx*qx + qy*qy + qz*qz);
  qw *= inv; qx *= inv; qy *= inv; qz *= inv;

  float R[3][3];
  R[0][0] = 1.f - 2.f*(qy*qy + qz*qz); R[0][1] = 2.f*(qx*qy - qw*qz); R[0][2] = 2.f*(qx*qz + qw*qy);
  R[1][0] = 2.f*(qx*qy + qw*qz); R[1][1] = 1.f - 2.f*(qx*qx + qz*qz); R[1][2] = 2.f*(qy*qz - qw*qx);
  R[2][0] = 2.f*(qx*qz - qw*qy); R[2][1] = 2.f*(qy*qz + qw*qx); R[2][2] = 1.f - 2.f*(qx*qx + qy*qy);

  float s0 = scales[g*3+0], s1 = scales[g*3+1], s2 = scales[g*3+2];
  float iv[3] = { 1.f/(s0*s0), 1.f/(s1*s1), 1.f/(s2*s2) };

  float A00 = 0.f, A01 = 0.f, A02 = 0.f, A11 = 0.f, A12 = 0.f, A22 = 0.f;
  #pragma unroll
  for (int k = 0; k < 3; ++k) {
    A00 += iv[k]*R[k][0]*R[k][0];
    A01 += iv[k]*R[k][0]*R[k][1];
    A02 += iv[k]*R[k][0]*R[k][2];
    A11 += iv[k]*R[k][1]*R[k][1];
    A12 += iv[k]*R[k][1]*R[k][2];
    A22 += iv[k]*R[k][2]*R[k][2];
  }

  float mx = means[g*3+0], my = means[g*3+1], mz = means[g*3+2];
  float Am0 = A00*mx + A01*my + A02*mz;
  float Am1 = A01*mx + A11*my + A12*mz;
  float Am2 = A02*mx + A12*my + A22*mz;
  float muAmu = mx*Am0 + my*Am1 + mz*Am2;

  const float s = -0.72134752044448170368f;   // -0.5 * log2(e)
  float* h = Hg + g*H_STRIDE;
  h[0] = s*A00;       h[1] = s*A11;       h[2] = s*A22;
  h[3] = s*2.f*A01;   h[4] = s*2.f*A02;   h[5] = s*2.f*A12;
  h[6] = s*-2.f*Am0;  h[7] = s*-2.f*Am1;  h[8] = s*-2.f*Am2;
  h[9] = s*muAmu + log2f(opac[g]);
  h[10] = 0.f; h[11] = 0.f;

  #pragma unroll
  for (int c = 0; c < 16; ++c)
    ST[c*N_GAUSS + g] = (c < N_CLASSES) ? (_Float16)sem[g*N_CLASSES + c] : (_Float16)0.f;
}

// ---------------------------------------------------------------------------
// Pass 2: fused evaluation + WMMA reduction, dual 16-point tiles per wave.
// Each wave owns 32 points (p0 tile and p0+16 tile). Per 32-gaussian chunk the
// h-vectors are loaded ONCE and reused for both tiles (halves L2 traffic,
// doubles ILP: two independent FMA chains per element), then two
// v_wmma_f32_16x16x32_f16 against the shared B operand.
// Layouts (wave32; C per ISA 7.12.2, A/B use a contiguous k order chosen here):
//   A: lane -> row m=lane%16; element e -> K = (lane>=16 ? 16:0) + e
//   B: lane -> col n=lane%16; elements 0..15 -> K = (lane>=16?16:0)+0..15
//      == 32 contiguous bytes of the class-transposed f16 semantics row.
//   C: lane L, vgpr r -> m = r + 8*(L>>4), n = L&15.
// ---------------------------------------------------------------------------
__global__ __launch_bounds__(256) void gauss_occ_kernel(
    const float* __restrict__ xyz, const float* __restrict__ Hg,
    const _Float16* __restrict__ ST, float* __restrict__ out)
{
  const int lane = threadIdx.x & 31;
  const int wave = threadIdx.x >> 5;
  const int tile = blockIdx.x * 8 + wave;           // 0..511 (dual tiles)
  const int m    = lane & 15;
  const int p0   = tile * 32 + m;
  const int p1   = p0 + 16;

  const float x0 = xyz[p0*3+0], y0 = xyz[p0*3+1], z0 = xyz[p0*3+2];
  const float x1 = xyz[p1*3+0], y1 = xyz[p1*3+1], z1 = xyz[p1*3+2];
  const float xx0 = x0*x0, yy0 = y0*y0, zz0 = z0*z0, xy0 = x0*y0, xz0 = x0*z0, yz0 = y0*z0;
  const float xx1 = x1*x1, yy1 = y1*y1, zz1 = z1*z1, xy1 = x1*y1, xz1 = x1*z1, yz1 = y1*z1;

  const int  kadd = (lane >= 16) ? 8  : 0;   // hi half-wave offset, used only by the prefetch
  const int  badd = (lane >= 16) ? 16 : 0;   // B-layout K offset for hi half-wave
  const _Float16* brow = ST + m * N_GAUSS;   // n == m == lane&15

  v8f acc0 = {};
  v8f acc1 = {};

  for (int k0 = 0; k0 < N_GAUSS; k0 += 32) {
    if (k0 + 32 < N_GAUSS)
      __builtin_prefetch(Hg + (k0 + 32 + kadd) * H_STRIDE, 0, 3);

    v16h a0, a1;
    #pragma unroll
    for (int e = 0; e < 16; ++e) {
      const int k = k0 + badd + e;
      const float* h = Hg + k * H_STRIDE;
      const float4 h0 = *(const float4*)(h);
      const float4 h1 = *(const float4*)(h + 4);
      const float2 h2 = *(const float2*)(h + 8);

      // Two split partial chains per point (depth ~5) -> 4 independent chains/elem.
      float qa0 = fmaf(h0.x, xx0, fmaf(h0.y, yy0, fmaf(h0.z, zz0, fmaf(h0.w, xy0, h2.y))));
      float qb0 = fmaf(h1.x, xz0, fmaf(h1.y, yz0, fmaf(h1.z, x0, fmaf(h1.w, y0, h2.x * z0))));
      float qa1 = fmaf(h0.x, xx1, fmaf(h0.y, yy1, fmaf(h0.z, zz1, fmaf(h0.w, xy1, h2.y))));
      float qb1 = fmaf(h1.x, xz1, fmaf(h1.y, yz1, fmaf(h1.z, x1, fmaf(h1.w, y1, h2.x * z1))));

      a0[e] = (_Float16)__builtin_amdgcn_exp2f(qa0 + qb0);   // w = opacity*exp(-q/2)
      a1[e] = (_Float16)__builtin_amdgcn_exp2f(qa1 + qb1);
    }

    const v16h b = *(const v16h*)(brow + k0 + badd);

    acc0 = __builtin_amdgcn_wmma_f32_16x16x32_f16(
        false, a0, false, b, (short)0, acc0, false, false);
    acc1 = __builtin_amdgcn_wmma_f32_16x16x32_f16(
        false, a1, false, b, (short)0, acc1, false, false);
  }

  // Scatter both C tiles: lane holds column n for rows r(+8 if hi half).
  const int n    = lane & 15;
  const int mrow = (lane >= 16) ? 8 : 0;
  if (n < N_CLASSES) {
    #pragma unroll
    for (int r = 0; r < 8; ++r) {
      const int pm0 = tile * 32 + mrow + r;
      out[pm0 * N_CLASSES + n]        = acc0[r];
      out[(pm0 + 16) * N_CLASSES + n] = acc1[r];
    }
  }
}

// ---------------------------------------------------------------------------
extern "C" void kernel_launch(void* const* d_in, const int* in_sizes, int n_in,
                              void* d_out, int out_size, void* d_ws, size_t ws_size,
                              hipStream_t stream) {
  const float* xyz       = (const float*)d_in[0];
  const float* means     = (const float*)d_in[1];
  const float* scales    = (const float*)d_in[2];
  const float* rotations = (const float*)d_in[3];
  const float* opacities = (const float*)d_in[4];
  const float* semantics = (const float*)d_in[5];
  float*       out       = (float*)d_out;

  float*     Hg = (float*)d_ws;                                   // 3200*12*4   = 153600 B
  _Float16*  ST = (_Float16*)((char*)d_ws + N_GAUSS*H_STRIDE*4);  // 16*3200*2   = 102400 B

  gauss_prep_kernel<<<(N_GAUSS + 255) / 256, 256, 0, stream>>>(
      means, scales, rotations, opacities, semantics, Hg, ST);

  gauss_occ_kernel<<<N_POINTS / (32 * 8), 256, 0, stream>>>(xyz, Hg, ST, out);
}
